// ViT_15358803051012
// MI455X (gfx1250) — matrix operand order repaired and checked by reading, not compiled, not run on hardware
//
#include <hip/hip_runtime.h>
#include <hip/hip_bf16.h>

// CDNA5 / gfx1250 ViT kernel, round 4 (final shape).
// - wave32 WMMA 16x16x32 f16->f32 for every matmul, including the fc1 head GEMM
// - pos_emb folded into per-(patch,out) bias (projection K: 32 -> 16)
// - full 32-lane B-fragments pre-baked in d_ws (zeros baked in) -> pure b128 loads
// - Q/K stored transposed [b][d][s] so attention fragments load as ds b128
// - softmax via width-16 / width-8 lane shuffles on the C-fragment layout
// - 256-thread blocks (8 waves): ~150 KB LDS/block, 2 blocks/WGP -> 4 waves/SIMD

typedef _Float16     v16h  __attribute__((ext_vector_type(16)));
typedef float        v8f   __attribute__((ext_vector_type(8)));
typedef unsigned int u32x4 __attribute__((ext_vector_type(4)));

union H16 { v16h h; u32x4 q[2]; };

#define NPAT   49
#define SPAD   64      // seq padded to 64 (zeros in [49,64))
#define TPAD   416     // 392 = 49*8 flat tok3, padded to 13*32 for fc1 WMMA K
#define NWAVES 8
#define BLOCK  256
#define BTOT   16384

__device__ __forceinline__ v8f wmma_f16(v16h a, v16h b, v8f c) {
  return __builtin_amdgcn_wmma_f32_16x16x32_f16(false, a, false, b, (short)0, c,
                                                false, false);
}

// ---------------------------------------------------------------------------
// Prep: full 32-lane WMMA B-fragments (f16) + folded pos_emb bias in d_ws.
// ---------------------------------------------------------------------------
__global__ void vit_prep_kernel(const float* __restrict__ pe,
                                const float* __restrict__ WQ1, const float* __restrict__ WK1,
                                const float* __restrict__ WV1,
                                const float* __restrict__ WQ2, const float* __restrict__ WK2,
                                const float* __restrict__ WV2,
                                const float* __restrict__ fc1w,
                                _Float16* __restrict__ B1, float* __restrict__ bias1,
                                _Float16* __restrict__ B2, float* __restrict__ bias2,
                                _Float16* __restrict__ fc1wh) {
  int t = blockIdx.x * blockDim.x + threadIdx.x;
  int gsz = (int)(blockDim.x * gridDim.x);
  if (t < NPAT * 32) {
    int s = t / 32, l = t - s * 32;
    const float* W1[3] = {WQ1, WK1, WV1};
    const float* W2[3] = {WQ2, WK2, WV2};
    for (int m = 0; m < 3; ++m) {
      _Float16* dst = B1 + (size_t)((m * NPAT + s) * 32 + l) * 16;
      if (l < 16) {
        const float* w = W1[m] + (size_t)(s * 16 + l) * 32;
        float bias = 0.f;
        for (int h = 0; h < 16; ++h) {
          dst[h] = (_Float16)w[h];
          bias = fmaf(pe[s * 16 + h], w[16 + h], bias);
        }
        bias1[(m * NPAT + s) * 16 + l] = bias;
        bias2[(m * NPAT + s) * 16 + l] = 0.f;
      } else {
        for (int h = 0; h < 16; ++h) dst[h] = (_Float16)0;
      }
      _Float16* dst2 = B2 + (size_t)((m * NPAT + s) * 32 + l) * 16;
      if (l < 8) {
        const float* w2 = W2[m] + (size_t)(s * 8 + l) * 16;
        for (int h = 0; h < 16; ++h) dst2[h] = (_Float16)w2[h];
      } else {
        for (int h = 0; h < 16; ++h) dst2[h] = (_Float16)0;
      }
    }
  }
  for (int idx = t; idx < 16 * TPAD; idx += gsz) {
    int j = idx / TPAD, h = idx - j * TPAD;
    fc1wh[idx] = (h < 392) ? (_Float16)fc1w[j * 392 + h] : (_Float16)0;
  }
}

// ---------------------------------------------------------------------------
// Per-patch projections. A: M=b(16), K=i(16 real, padded). B from d_ws.
// Q,K results stored transposed [b][d][s64]; V stored [b][s64][e].
// ---------------------------------------------------------------------------
__device__ __forceinline__ void project_layer(
    const _Float16* sTok,                        // LDS [16][49][16]
    const _Float16* __restrict__ Bfr,            // global [3][49][32][16] f16
    const float*    __restrict__ bias,           // global [3][49][16]
    _Float16* sQt, _Float16* sKt, _Float16* sV,  // LDS: Qt/Kt [16][16][64], V [16][64][16]
    int wave, int lane) {
  const int l15 = lane & 15, hi = lane >> 4;
  for (int s = wave; s < NPAT; s += NWAVES) {
    // A fragment: lanes<16 hold K=0..15 contiguous (two ds b128), lanes>=16 (K=16..31) zero
    H16 a;
    a.q[0] = hi ? (u32x4){0u, 0u, 0u, 0u} : *(const u32x4*)(sTok + (l15 * NPAT + s) * 16);
    a.q[1] = hi ? (u32x4){0u, 0u, 0u, 0u} : *(const u32x4*)(sTok + (l15 * NPAT + s) * 16 + 8);
#pragma unroll
    for (int m = 0; m < 3; ++m) {
      // B fragment: two unconditional coalesced global b128 loads
      const u32x4* bp = (const u32x4*)(Bfr + (size_t)((m * NPAT + s) * 32 + lane) * 16);
      H16 bf;
      bf.q[0] = bp[0];
      bf.q[1] = bp[1];
      float cb = bias[(m * NPAT + s) * 16 + l15];  // C col N=l15 (branchless)
      v8f c;
#pragma unroll
      for (int r = 0; r < 8; ++r) c[r] = cb;
      c = wmma_f16(a.h, bf.h, c);
      // D: (M=b=r+8hi, N=o=l15)
#pragma unroll
      for (int r = 0; r < 8; ++r) {
        int b = r + 8 * hi;
        _Float16 d = (_Float16)c[r];
        if (m == 0)      sQt[(b * 16 + l15) * SPAD + s] = d;   // [b][d][s]
        else if (m == 1) sKt[(b * 16 + l15) * SPAD + s] = d;   // [b][e][s]
        else             sV[(b * SPAD + s) * 16 + l15] = d;    // [b][s][e]
      }
    }
  }
}

// ---------------------------------------------------------------------------
// Attention + softmax; images distributed across waves.
// S[d][e] = (1/7) sum_s Q[b][s][d]K[b][s][e]   (K=64, 2 WMMAs)
// out[s][d] = sum_e S[d][e]V[b][s][e]          (4 s-tiles, B = S^T)
// MODE 0: softmax over 16 dims -> f16 [16][49][16]
// MODE 1: softmax over 8 dims  -> f16 tok3 flat [16][TPAD] (i = s*8+d)
// ---------------------------------------------------------------------------
template <int MODE>
__device__ __forceinline__ void attention(
    const _Float16* sQt, const _Float16* sKt,  // LDS [16][16][64]
    const _Float16* sV,                        // LDS [16][64][16]
    _Float16* sS,                              // LDS per-wave [16][16]
    _Float16* dstTok, int wave, int lane) {
  const int l15 = lane & 15, hi = lane >> 4;
  const unsigned ml = hi ? 0u : 0xffffffffu;
  const u32x4 msk = {ml, ml, ml, ml};
  for (int b = wave; b < 16; b += NWAVES) {
    const _Float16* Qr = sQt + (b * 16 + l15) * SPAD;  // row d=l15 over s
    const _Float16* Kr = sKt + (b * 16 + l15) * SPAD;  // row e=l15 over s
    const _Float16* Vb = sV + b * SPAD * 16;
    v8f c;
#pragma unroll
    for (int r = 0; r < 8; ++r) c[r] = 0.f;
#pragma unroll
    for (int s0 = 0; s0 < 64; s0 += 32) {
      // A 16x32 f16 K-map: halves0-7 -> s0+8hi+h ; halves8-15 -> s0+16+8hi+h
      H16 qa;
      qa.q[0] = *(const u32x4*)(Qr + s0 + 8 * hi);
      qa.q[1] = *(const u32x4*)(Qr + s0 + 16 + 8 * hi);
      // B 32x16 f16 K-map: halves0-7 -> s0+8hi+h ; halves8-15 -> s0+16+8hi+h
      H16 kb;
      kb.q[0] = *(const u32x4*)(Kr + s0 + 8 * hi);
      kb.q[1] = *(const u32x4*)(Kr + s0 + 16 + 8 * hi);
      c = wmma_f16(qa.h, kb.h, c);
    }
    // S*(1/7) -> per-wave LDS; lane<16 then holds row l15 = B-frag of S^T
#pragma unroll
    for (int r = 0; r < 8; ++r)
      sS[(r + 8 * hi) * 16 + l15] = (_Float16)(c[r] * 0.14285714285714285f);
    H16 sb;
    sb.q[0] = (*(const u32x4*)(sS + l15 * 16)) & msk;
    sb.q[1] = (*(const u32x4*)(sS + l15 * 16 + 8)) & msk;

#pragma unroll
    for (int s0 = 0; s0 < 64; s0 += 16) {
      // A[s'=l15][K=e]: lanes<16 hold e=0..15 contiguous, lanes>=16 (e>=16) zero
      H16 va;
      va.q[0] = (*(const u32x4*)(Vb + (s0 + l15) * 16)) & msk;
      va.q[1] = (*(const u32x4*)(Vb + (s0 + l15) * 16 + 8)) & msk;
      v8f co;
#pragma unroll
      for (int r = 0; r < 8; ++r) co[r] = 0.f;
      co = wmma_f16(va.h, sb.h, co);
      // row s = s0 + r + 8hi spans 16 lanes (d=l15): shuffle softmax
#pragma unroll
      for (int r = 0; r < 8; ++r) {
        float v = co[r];
        float mx = v;
        if (MODE == 0) {
          mx = fmaxf(mx, __shfl_xor(mx, 8, 16));
          mx = fmaxf(mx, __shfl_xor(mx, 4, 16));
          mx = fmaxf(mx, __shfl_xor(mx, 2, 16));
          mx = fmaxf(mx, __shfl_xor(mx, 1, 16));
        } else {
          mx = fmaxf(mx, __shfl_xor(mx, 4, 8));
          mx = fmaxf(mx, __shfl_xor(mx, 2, 8));
          mx = fmaxf(mx, __shfl_xor(mx, 1, 8));
        }
        float e = __expf(v - mx);
        float sm = e;
        if (MODE == 0) {
          sm += __shfl_xor(sm, 8, 16);
          sm += __shfl_xor(sm, 4, 16);
          sm += __shfl_xor(sm, 2, 16);
          sm += __shfl_xor(sm, 1, 16);
        } else {
          sm += __shfl_xor(sm, 4, 8);
          sm += __shfl_xor(sm, 2, 8);
          sm += __shfl_xor(sm, 1, 8);
        }
        float p = e / sm;
        int s = s0 + r + 8 * hi;
        if (MODE == 0) {
          if (s < NPAT) dstTok[(b * NPAT + s) * 16 + l15] = (_Float16)p;
        } else {
          if (s < NPAT && l15 < 8) dstTok[b * TPAD + s * 8 + l15] = (_Float16)p;
        }
      }
    }
  }
}

// ---------------------------------------------------------------------------
// Main kernel: one block = 8 waves = one 16-image tile.
// ---------------------------------------------------------------------------
__global__ void __launch_bounds__(BLOCK)
vit_main_kernel(const float* __restrict__ X, float* __restrict__ Out,
                const _Float16* __restrict__ B1, const float* __restrict__ bias1,
                const _Float16* __restrict__ B2, const float* __restrict__ bias2,
                const _Float16* __restrict__ fc1wh, const float* __restrict__ fc1b,
                const float* __restrict__ fc2w, const float* __restrict__ fc2b) {
  __shared__ __align__(16) _Float16 sQt[16 * 16 * SPAD];   // 32 KB [b][d][s]
  __shared__ __align__(16) _Float16 sKt[16 * 16 * SPAD];   // 32 KB [b][e][s]
  __shared__ __align__(16) _Float16 sV [16 * SPAD * 16];   // 32 KB [b][s][e]
  __shared__ __align__(16) _Float16 sT [16 * NPAT * 16];   // 25 KB patches -> tok2
  __shared__ __align__(16) _Float16 sS [NWAVES][16 * 16];  // 4 KB per-wave S
  __shared__ __align__(16) _Float16 sT3[16 * TPAD];        // 13 KB tok3 flat, padded
  __shared__ float sHp[NWAVES][16 * 16];                   // 8 KB fc1 partials
  __shared__ float sH [16][16];                            // 1 KB fc1 activations

  const int tid = threadIdx.x;
  const int lane = tid & 31, wave = tid >> 5;
  const int l15 = lane & 15, hi = lane >> 4;
  const int b0 = blockIdx.x * 16;

  // Zero padding regions
  for (int idx = tid; idx < 16 * 16 * (SPAD - NPAT); idx += BLOCK) {
    int bd = idx / (SPAD - NPAT);
    int s = NPAT + (idx - bd * (SPAD - NPAT));
    sQt[bd * SPAD + s] = (_Float16)0;
    sKt[bd * SPAD + s] = (_Float16)0;
  }
  for (int idx = tid; idx < 16 * (SPAD - NPAT) * 16; idx += BLOCK) {
    int bl = idx / ((SPAD - NPAT) * 16);
    int rem = idx - bl * ((SPAD - NPAT) * 16);
    sV[(bl * SPAD + NPAT + rem / 16) * 16 + (rem & 15)] = (_Float16)0;
  }
  for (int idx = tid; idx < 16 * (TPAD - 392); idx += BLOCK) {
    int bl = idx / (TPAD - 392);
    sT3[bl * TPAD + 392 + (idx - bl * (TPAD - 392))] = (_Float16)0;
  }

  // Phase 0: grayscale + patchify, 4 pixels/thread (float4 loads, packed LDS store)
  for (int p = tid; p < 16 * 196; p += BLOCK) {
    int bl = p / 196, k = p - bl * 196;
    int pix = k * 4;
    const float4* px = (const float4*)(X + (size_t)(b0 + bl) * 3 * 784 + pix);
    float4 r = px[0], g = px[196], bch = px[392];
    int y = pix / 28, xx = pix - y * 28;
    int s = (y >> 2) * 7 + (xx >> 2);
    int i = (y & 3) * 4;
    _Float16* dst = sT + (bl * NPAT + s) * 16 + i;
    dst[0] = (_Float16)(0.299f * r.x + 0.587f * g.x + 0.114f * bch.x);
    dst[1] = (_Float16)(0.299f * r.y + 0.587f * g.y + 0.114f * bch.y);
    dst[2] = (_Float16)(0.299f * r.z + 0.587f * g.z + 0.114f * bch.z);
    dst[3] = (_Float16)(0.299f * r.w + 0.587f * g.w + 0.114f * bch.w);
  }
  __syncthreads();

  project_layer(sT, B1, bias1, sQt, sKt, sV, wave, lane);    // 147 WMMAs
  __syncthreads();
  attention<0>(sQt, sKt, sV, &sS[wave][0], sT, wave, lane);  // 96 WMMAs -> tok2
  __syncthreads();
  project_layer(sT, B2, bias2, sQt, sKt, sV, wave, lane);    // 147 WMMAs
  __syncthreads();
  attention<1>(sQt, sKt, sV, &sS[wave][0], sT3, wave, lane); // 96 WMMAs -> tok3
  __syncthreads();

  // fc1 as WMMA: h = relu(tok3[16x392] @ fc1w^T[392x16] + b). K split across waves.
  {
    v8f c;
#pragma unroll
    for (int r = 0; r < 8; ++r) c[r] = 0.f;
    const _Float16* arow = sT3 + l15 * TPAD;       // A row M=b=l15
    const _Float16* brow = fc1wh + l15 * TPAD;     // B col N=j=l15
    for (int k0 = wave * 32; k0 < TPAD; k0 += NWAVES * 32) {  // 13 chunks total
      H16 a, bf;
      a.q[0] = *(const u32x4*)(arow + k0 + 8 * hi);
      a.q[1] = *(const u32x4*)(arow + k0 + 16 + 8 * hi);
      bf.q[0] = *(const u32x4*)(brow + k0 + 8 * hi);
      bf.q[1] = *(const u32x4*)(brow + k0 + 16 + 8 * hi);
      c = wmma_f16(a.h, bf.h, c);
    }
#pragma unroll
    for (int r = 0; r < 8; ++r) sHp[wave][(r + 8 * hi) * 16 + l15] = c[r];
  }
  __syncthreads();
  for (int e = tid; e < 256; e += BLOCK) {  // reduce partials + bias + relu
    int j = e & 15;
    float sum = fc1b[j];
#pragma unroll
    for (int w = 0; w < NWAVES; ++w) sum += sHp[w][e];
    sH[e >> 4][j] = fmaxf(sum, 0.f);
  }
  __syncthreads();

  // fc2 (16->10) + softmax, one thread per image
  if (tid < 16) {
    int tb = tid;
    float z[10];
    float mx = -3.0e38f;
#pragma unroll
    for (int k = 0; k < 10; ++k) {
      float acc = fc2b[k];
#pragma unroll
      for (int j = 0; j < 16; ++j) acc = fmaf(sH[tb][j], fc2w[k * 16 + j], acc);
      z[k] = acc;
      mx = fmaxf(mx, acc);
    }
    float sm = 0.f;
#pragma unroll
    for (int k = 0; k < 10; ++k) { z[k] = __expf(z[k] - mx); sm += z[k]; }
    float inv = 1.f / sm;
    float* o = Out + (size_t)(b0 + tb) * 10;
#pragma unroll
    for (int k = 0; k < 10; ++k) o[k] = z[k] * inv;
  }
}

// ---------------------------------------------------------------------------
// d_ws layout (needs 333,184 B):
//   [0      , 150528)  B1   : 3 x 49 x 32 x 16 f16 (full fragments, zeros baked)
//   [150528 , 159936)  bias1: 3 x 49 x 16 f32 (pos_emb folded)
//   [159936 , 310464)  B2   : 3 x 49 x 32 x 16 f16
//   [310464 , 319872)  bias2: 3 x 49 x 16 f32 (zeros)
//   [319872 , 333184)  fc1wh: 16 x 416 f16 (padded)
// ---------------------------------------------------------------------------
extern "C" void kernel_launch(void* const* d_in, const int* in_sizes, int n_in,
                              void* d_out, int out_size, void* d_ws, size_t ws_size,
                              hipStream_t stream) {
  const float* x    = (const float*)d_in[0];
  const float* pe   = (const float*)d_in[1];
  const float* WQ1  = (const float*)d_in[2];
  const float* WK1  = (const float*)d_in[3];
  const float* WV1  = (const float*)d_in[4];
  const float* WQ2  = (const float*)d_in[5];
  const float* WK2  = (const float*)d_in[6];
  const float* WV2  = (const float*)d_in[7];
  const float* fc1w = (const float*)d_in[8];
  const float* fc1b = (const float*)d_in[9];
  const float* fc2w = (const float*)d_in[10];
  const float* fc2b = (const float*)d_in[11];
  float* out = (float*)d_out;

  char* ws = (char*)d_ws;
  _Float16* B1    = (_Float16*)(ws);
  float*    bias1 = (float*)(ws + 150528);
  _Float16* B2    = (_Float16*)(ws + 159936);
  float*    bias2 = (float*)(ws + 310464);
  _Float16* fc1wh = (_Float16*)(ws + 319872);

  vit_prep_kernel<<<(NPAT * 32 + 127) / 128, 128, 0, stream>>>(
      pe, WQ1, WK1, WV1, WQ2, WK2, WV2, fc1w, B1, bias1, B2, bias2, fc1wh);

  vit_main_kernel<<<BTOT / 16, BLOCK, 0, stream>>>(
      x, out, B1, bias1, B2, bias2, fc1wh, fc1b, fc2w, fc2b);
}
